// BevSpatialCrossAtten_82884278879187
// MI455X (gfx1250) — matrix operand order repaired and checked by reading, not compiled, not run on hardware
//
#include <hip/hip_runtime.h>
#include <hip/hip_bf16.h>

// ---- problem constants (from reference) ----
#define CAMS 6
#define HDS  8
#define LVLS 4
#define REFS 4
#define CDIM 256
#define DH   32     // CDIM / HDS

typedef __attribute__((ext_vector_type(16))) __bf16 v16bf;
typedef __attribute__((ext_vector_type(8)))  float  v8f;
typedef __attribute__((ext_vector_type(4)))  unsigned int u32x4;
typedef __attribute__((ext_vector_type(8)))  int          i32x8;
typedef __attribute__((ext_vector_type(4)))  int          i32x4;

// ============================================================
// Generic bf16 GEMM: D = A(M,K) * B(K,N) + bias, B given transposed (N,K).
// Block = 4 waves, all sharing one 16-column tile (tn). The 16xK B^T tile
// (8 KB for K=256) is staged into LDS once per block by the Tensor Data
// Mover, then each wave computes a 32x16 output tile (2 accumulators) with
// v_wmma_f32_16x16x32_bf16, reusing the LDS B fragment for both.
// Optional f32 residual add; output f32 and/or bf16.
// ============================================================
__global__ __launch_bounds__(128)
void gemm_bf16_tile(const __hip_bfloat16* __restrict__ A,
                    const __hip_bfloat16* __restrict__ BT,
                    const float* __restrict__ bias,
                    float* __restrict__ Df,
                    __hip_bfloat16* __restrict__ Dh,
                    const float* __restrict__ addsrc,
                    int M, int N, int K)
{
    extern __shared__ __hip_bfloat16 Bs[];   // 16 * K bf16 (dynamic LDS, base offset 0)
    const int lane = threadIdx.x & 31;
    const int wave = threadIdx.x >> 5;
    const int tn = blockIdx.y;

    // ---- stage B^T tile (rows tn*16 .. tn*16+15, all K) into LDS via TDM ----
    if (wave == 0) {
        unsigned long long gaddr =
            (unsigned long long)(const void*)(BT + (size_t)tn * 16 * (size_t)K);
        u32x4 g0;
        g0[0] = 1u;                                   // count=1 (valid user D#)
        g0[1] = 0u;                                   // lds_addr = 0
        g0[2] = (unsigned)(gaddr & 0xffffffffu);      // global_addr[31:0]
        g0[3] = (unsigned)((gaddr >> 32) & 0x01ffffffu) | (2u << 30); // addr[56:32] | type=2
        i32x8 g1;
        g1[0] = (int)(1u << 16);                      // workgroup_mask=0, data_size=1 (2B)
        g1[1] = (int)(((unsigned)K & 0xffffu) << 16); // tensor_dim0[15:0] in [31:16]
        g1[2] = (int)((((unsigned)K >> 16) & 0xffffu) | (16u << 16)); // dim0[31:16] | dim1[15:0]=16
        g1[3] = (int)(((unsigned)K & 0xffffu) << 16); // dim1[31:16]=0 | tile_dim0[15:0]=K
        g1[4] = 16;                                   // tile_dim1=16, tile_dim2=0
        g1[5] = K;                                    // tensor_dim0_stride[31:0]
        g1[6] = 0;                                    // stride0[47:32] | stride1[15:0]
        g1[7] = 0;                                    // stride1[47:16]
        i32x4 gz = {0, 0, 0, 0};
#if defined(__HIP_DEVICE_COMPILE__)
#if __clang_major__ >= 23
        i32x8 gz8 = {0, 0, 0, 0, 0, 0, 0, 0};
        __builtin_amdgcn_tensor_load_to_lds(g0, g1, gz, gz, gz8, 0);
#else
        __builtin_amdgcn_tensor_load_to_lds(g0, g1, gz, gz, 0);
#endif
        __builtin_amdgcn_s_wait_tensorcnt(0);
#endif
        (void)g0; (void)g1; (void)gz;
    }
    __syncthreads();

    const int half = lane >> 4;   // 0 or 1
    const int l16  = lane & 15;

    const int tm0 = (blockIdx.x * 4 + wave) * 2;   // two 16-row tiles per wave
    const int tm1 = tm0 + 1;
    if (tm0 * 16 >= M) return;
    const bool has1 = (tm1 * 16) < M;

    // A fragment: row = l16, K groups {half*8 .. +7} and {16+half*8 .. +7}
    // B fragment (from LDS): row = l16 (output col), K groups {half*8 .. +7} and {16+half*8 .. +7}
    const int ka = half * 8;
    const int kb = half * 8;
    const size_t arow0 = (size_t)(tm0 * 16 + l16) * (size_t)K;
    int r1 = tm1 * 16 + l16; if (r1 >= M) r1 = M - 1;   // clamp for safe loads
    const size_t arow1 = (size_t)r1 * (size_t)K;
    const __hip_bfloat16* Bl = Bs + (size_t)l16 * (size_t)K;

    v8f c0, c1;
    const float bv = bias ? bias[tn * 16 + l16] : 0.0f;
#pragma unroll
    for (int i = 0; i < 8; ++i) { c0[i] = bv; c1[i] = bv; }

    union Frag { uint4 q[2]; v16bf v; };

    for (int k = 0; k < K; k += 32) {
        Frag fb, fa0, fa1;
        fb.q[0]  = *(const uint4*)(Bl + k + kb);           // ds_load_b128
        fb.q[1]  = *(const uint4*)(Bl + k + 16 + kb);
        fa0.q[0] = *(const uint4*)(A + arow0 + k + ka);
        fa0.q[1] = *(const uint4*)(A + arow0 + k + 16 + ka);
        fa1.q[0] = *(const uint4*)(A + arow1 + k + ka);
        fa1.q[1] = *(const uint4*)(A + arow1 + k + 16 + ka);
#if defined(__HIP_DEVICE_COMPILE__)
        c0 = __builtin_amdgcn_wmma_f32_16x16x32_bf16(
                false, fa0.v, false, fb.v, (short)0, c0, false, false);
        c1 = __builtin_amdgcn_wmma_f32_16x16x32_bf16(
                false, fa1.v, false, fb.v, (short)0, c1, false, false);
#endif
    }

    // C/D layout: col = l16, rows = 8*half + i
    const int n = tn * 16 + l16;
#pragma unroll
    for (int i = 0; i < 8; ++i) {
        size_t idx = (size_t)(tm0 * 16 + 8 * half + i) * (size_t)N + n;
        float x = c0[i];
        if (addsrc) x += addsrc[idx];
        if (Df) Df[idx] = x;
        if (Dh) Dh[idx] = __float2bfloat16(x);
    }
    if (has1) {
#pragma unroll
        for (int i = 0; i < 8; ++i) {
            size_t idx = (size_t)(tm1 * 16 + 8 * half + i) * (size_t)N + n;
            float x = c1[i];
            if (addsrc) x += addsrc[idx];
            if (Df) Df[idx] = x;
            if (Dh) Dh[idx] = __float2bfloat16(x);
        }
    }
}

// ============================================================
// Weight transpose + cast: W(K,N) f32 -> WT(N,K) bf16
// ============================================================
__global__ void transpose_cast_kernel(const float* __restrict__ W,
                                      __hip_bfloat16* __restrict__ WT,
                                      int K, int N)
{
    int idx = blockIdx.x * blockDim.x + threadIdx.x;
    if (idx >= K * N) return;
    int k = idx / N, n = idx - k * N;
    WT[(size_t)n * K + k] = __float2bfloat16(W[idx]);
}

// q + query_pos -> bf16
__global__ void qsum_kernel(const float* __restrict__ q,
                            const float* __restrict__ qp,
                            __hip_bfloat16* __restrict__ out, int n)
{
    int i = blockIdx.x * blockDim.x + threadIdx.x;
    if (i < n) out[i] = __float2bfloat16(q[i] + qp[i]);
}

// value (NK, CAMS, C) f32 -> (CAMS, NK, C) bf16
__global__ void valcast_kernel(const float* __restrict__ val,
                               __hip_bfloat16* __restrict__ out, int NK)
{
    int i = blockIdx.x * blockDim.x + threadIdx.x;
    int total = NK * CAMS * CDIM;
    if (i >= total) return;
    int c = i % CDIM;
    int t = i / CDIM;
    int k = t % NK;
    int cam = t / NK;
    out[i] = __float2bfloat16(val[((size_t)k * CAMS + cam) * CDIM + c]);
}

// Project reference points into cameras: uv + mask per (cam,q,ref)
__global__ void project_kernel(const float* __restrict__ brp,
                               const float* __restrict__ l2i,
                               const float* __restrict__ img_shape,
                               float2* __restrict__ refuv,
                               float* __restrict__ maskf, int NQ)
{
    int i = blockIdx.x * blockDim.x + threadIdx.x;
    int total = CAMS * NQ * REFS;
    if (i >= total) return;
    int r = i % REFS;
    int t = i / REFS;
    int q = t % NQ;
    int cam = t / NQ;

    const float* rp = brp + ((size_t)q * REFS + r) * 3;
    float X = rp[0] * 102.4f - 51.2f;   // PC_RANGE x
    float Y = rp[1] * 102.4f - 51.2f;   // PC_RANGE y
    float Z = rp[2] * 8.0f   - 5.0f;    // PC_RANGE z

    const float* M = l2i + (size_t)cam * 12;  // (3,4) row-major
    float cx = M[0] * X + M[1] * Y + M[2]  * Z + M[3];
    float cy = M[4] * X + M[5] * Y + M[6]  * Z + M[7];
    float cz = M[8] * X + M[9] * Y + M[10] * Z + M[11];

    float d  = fmaxf(cz, 1e-5f);
    float iw = img_shape[cam * 2 + 1];  // W (reversed last axis in ref)
    float ih = img_shape[cam * 2 + 0];  // H
    float u = cx / d / iw;
    float v = cy / d / ih;
    bool m = (cz > 1e-5f) && (u > 0.f) && (u < 1.f) && (v > 0.f) && (v < 1.f);
    refuv[i] = make_float2(u, v);
    maskf[i] = m ? 1.0f : 0.0f;
}

// valid (any over refs) per (cam,q); count per q
__global__ void valid_count_kernel(const float* __restrict__ maskf,
                                   float* __restrict__ validf,
                                   float* __restrict__ countf, int NQ)
{
    int q = blockIdx.x * blockDim.x + threadIdx.x;
    if (q >= NQ) return;
    float cnt = 0.f;
#pragma unroll
    for (int cam = 0; cam < CAMS; ++cam) {
        const float* m = maskf + ((size_t)cam * NQ + q) * REFS;
        float v = fmaxf(fmaxf(m[0], m[1]), fmaxf(m[2], m[3]));
        validf[cam * NQ + q] = v;
        cnt += v;
    }
    countf[q] = fmaxf(cnt, 1.0f);
}

// softmax over 16 (L*R*P) values per (q, head), in place
__global__ void softmax_kernel(float* __restrict__ logits, int NQ)
{
    int i = blockIdx.x * blockDim.x + threadIdx.x;
    if (i >= NQ * HDS) return;
    float* p = logits + (size_t)i * 16;   // (q*HDS + h)*16 == q*128 + h*16
    float mx = p[0];
#pragma unroll
    for (int j = 1; j < 16; ++j) mx = fmaxf(mx, p[j]);
    float e[16], s = 0.f;
#pragma unroll
    for (int j = 0; j < 16; ++j) { e[j] = __expf(p[j] - mx); s += e[j]; }
    float inv = 1.0f / s;
#pragma unroll
    for (int j = 0; j < 16; ++j) p[j] = e[j] * inv;
}

// ============================================================
// Deformable bilinear sampling. One wave per (cam, q, head);
// lane = channel d in [0,32). Corner gathers are coalesced 64B bf16 reads.
// ============================================================
__global__ __launch_bounds__(256)
void sample_kernel(const __hip_bfloat16* __restrict__ vproj,  // (CAMS, NK, 256) bf16
                   const float2* __restrict__ refuv,          // (CAMS, NQ, REFS)
                   const float* __restrict__ off,             // (NQ, 256)
                   const float* __restrict__ aw,              // (NQ, 128)
                   const int* __restrict__ shapes,            // (4,2)
                   const int* __restrict__ lsi,               // (4,)
                   __hip_bfloat16* __restrict__ attn_in,      // (CAMS*NQ, 256) bf16
                   int NQ, int NK)
{
    const int lane = threadIdx.x & 31;
    const int gid = (blockIdx.x * blockDim.x + threadIdx.x) >> 5;
    const int total = CAMS * NQ * HDS;
    if (gid >= total) return;
    const int h = gid % HDS;
    const int t = gid / HDS;
    const int q = t % NQ;
    const int cam = t / NQ;

    const float2* uv  = refuv + ((size_t)cam * NQ + q) * REFS;
    const float* offq = off + (size_t)q * 256 + h * 32;   // (h, l, r, 2)
    const float* awq  = aw  + (size_t)q * 128 + h * 16;   // (h, l*4+r)
    const __hip_bfloat16* vcam = vproj + (size_t)cam * NK * CDIM + h * DH + lane;

    float acc = 0.0f;
    for (int l = 0; l < LVLS; ++l) {
        const int Hl = shapes[2 * l], Wl = shapes[2 * l + 1];
        const int s = lsi[l];
        const __hip_bfloat16* vlev = vcam + (size_t)s * CDIM;
#pragma unroll
        for (int r = 0; r < REFS; ++r) {
            float2 p = uv[r];
            float ox = offq[(l * REFS + r) * 2 + 0];
            float oy = offq[(l * REFS + r) * 2 + 1];
            float w  = awq[l * REFS + r];
            // loc = ref + off/norm  ->  x = loc.x*Wl - 0.5
            float x = p.x * (float)Wl + ox - 0.5f;
            float y = p.y * (float)Hl + oy - 0.5f;
            float x0 = floorf(x), y0 = floorf(y);
            float lx = x - x0, ly = y - y0;
            int ix = (int)x0, iy = (int)y0;
            float w00 = (1.f - ly) * (1.f - lx) * w;
            float w01 = (1.f - ly) * lx * w;
            float w10 = ly * (1.f - lx) * w;
            float w11 = ly * lx * w;
#pragma unroll
            for (int cy = 0; cy < 2; ++cy)
#pragma unroll
            for (int cx = 0; cx < 2; ++cx) {
                int yy = iy + cy, xx = ix + cx;
                float cw = cy ? (cx ? w11 : w10) : (cx ? w01 : w00);
                if (yy >= 0 && yy < Hl && xx >= 0 && xx < Wl) {
                    acc += cw * __bfloat162float(vlev[((size_t)yy * Wl + xx) * CDIM]);
                }
            }
        }
    }
    attn_in[((size_t)cam * NQ + q) * CDIM + h * DH + lane] = __float2bfloat16(acc);
}

// masked mean over cameras -> slots bf16
__global__ void reduce_slots_kernel(const float* __restrict__ attn_out,  // (CAMS*NQ, 256)
                                    const float* __restrict__ validf,    // (CAMS, NQ)
                                    const float* __restrict__ countf,    // (NQ,)
                                    __hip_bfloat16* __restrict__ slots,  // (NQ, 256)
                                    int NQ)
{
    int i = blockIdx.x * blockDim.x + threadIdx.x;
    if (i >= NQ * CDIM) return;
    int c = i % CDIM;
    int q = i / CDIM;
    float s = 0.f;
#pragma unroll
    for (int cam = 0; cam < CAMS; ++cam)
        s += attn_out[((size_t)cam * NQ + q) * CDIM + c] * validf[cam * NQ + q];
    slots[i] = __float2bfloat16(s / countf[q]);
}

// ============================================================
// host launcher
// ============================================================
extern "C" void kernel_launch(void* const* d_in, const int* in_sizes, int n_in,
                              void* d_out, int out_size, void* d_ws, size_t ws_size,
                              hipStream_t stream)
{
    const float* query     = (const float*)d_in[0];
    const float* query_pos = (const float*)d_in[1];
    const float* value     = (const float*)d_in[2];
    const float* brp       = (const float*)d_in[3];
    const float* l2i       = (const float*)d_in[4];
    const float* img_shape = (const float*)d_in[5];
    const int*   shapes    = (const int*)d_in[6];
    const int*   lsi       = (const int*)d_in[7];
    const float* W_val  = (const float*)d_in[8];
    const float* b_val  = (const float*)d_in[9];
    const float* W_samp = (const float*)d_in[10];
    const float* b_samp = (const float*)d_in[11];
    const float* W_attn = (const float*)d_in[12];
    const float* b_attn = (const float*)d_in[13];
    const float* W_dout = (const float*)d_in[14];
    const float* b_dout = (const float*)d_in[15];
    const float* W_out  = (const float*)d_in[16];
    const float* b_out  = (const float*)d_in[17];

    const int NQ = in_sizes[0] / CDIM;                 // 10000
    const int NK = in_sizes[2] / (CAMS * CDIM);        // 13600
    const int MQ = NQ;                                 // query rows
    const int MV = CAMS * NK;                          // value rows
    const int MA = CAMS * NQ;                          // attn rows
    const size_t LDSB = 16 * CDIM * sizeof(__hip_bfloat16);  // 8 KB B tile

    // ---- workspace carve-out ----
    char* w = (char*)d_ws;
    auto alloc = [&](size_t bytes) -> char* {
        char* p = w; w += (bytes + 255) & ~(size_t)255; return p;
    };
    __hip_bfloat16* WT_val  = (__hip_bfloat16*)alloc((size_t)CDIM * CDIM * 2);
    __hip_bfloat16* WT_samp = (__hip_bfloat16*)alloc((size_t)CDIM * CDIM * 2);
    __hip_bfloat16* WT_attn = (__hip_bfloat16*)alloc((size_t)128 * CDIM * 2);
    __hip_bfloat16* WT_dout = (__hip_bfloat16*)alloc((size_t)CDIM * CDIM * 2);
    __hip_bfloat16* WT_out  = (__hip_bfloat16*)alloc((size_t)CDIM * CDIM * 2);
    __hip_bfloat16* qsum    = (__hip_bfloat16*)alloc((size_t)MQ * CDIM * 2);
    __hip_bfloat16* val_bf  = (__hip_bfloat16*)alloc((size_t)MV * CDIM * 2);
    __hip_bfloat16* vproj   = (__hip_bfloat16*)alloc((size_t)MV * CDIM * 2);
    float*          off_f   = (float*)alloc((size_t)MQ * 256 * 4);
    float*          aw_f    = (float*)alloc((size_t)MQ * 128 * 4);
    float2*         refuv   = (float2*)alloc((size_t)CAMS * NQ * REFS * 8);
    float*          maskf   = (float*)alloc((size_t)CAMS * NQ * REFS * 4);
    float*          validf  = (float*)alloc((size_t)CAMS * NQ * 4);
    float*          countf  = (float*)alloc((size_t)NQ * 4);
    __hip_bfloat16* attn_in = (__hip_bfloat16*)alloc((size_t)MA * CDIM * 2);
    float*          attn_out= (float*)alloc((size_t)MA * CDIM * 4);
    __hip_bfloat16* slots   = (__hip_bfloat16*)alloc((size_t)MQ * CDIM * 2);

    const int T = 256;
    auto grid1 = [&](int n) { return dim3((unsigned)((n + T - 1) / T)); };
    auto gemm_grid = [&](int M, int N) {
        // 4 waves/block, each wave does 2 row-tiles => 8 row-tiles per block
        return dim3((unsigned)((M / 16 + 7) / 8), (unsigned)(N / 16));
    };

    // 1) weight transposes (f32 -> bf16 (N,K))
    transpose_cast_kernel<<<grid1(CDIM * CDIM), T, 0, stream>>>(W_val,  WT_val,  CDIM, CDIM);
    transpose_cast_kernel<<<grid1(CDIM * CDIM), T, 0, stream>>>(W_samp, WT_samp, CDIM, CDIM);
    transpose_cast_kernel<<<grid1(CDIM * 128),  T, 0, stream>>>(W_attn, WT_attn, CDIM, 128);
    transpose_cast_kernel<<<grid1(CDIM * CDIM), T, 0, stream>>>(W_dout, WT_dout, CDIM, CDIM);
    transpose_cast_kernel<<<grid1(CDIM * CDIM), T, 0, stream>>>(W_out,  WT_out,  CDIM, CDIM);

    // 2) q + pos -> bf16 ; value transpose -> bf16
    qsum_kernel<<<grid1(MQ * CDIM), T, 0, stream>>>(query, query_pos, qsum, MQ * CDIM);
    valcast_kernel<<<grid1(MV * CDIM), T, 0, stream>>>(value, val_bf, NK);

    // 3) camera projection + masks
    project_kernel<<<grid1(CAMS * NQ * REFS), T, 0, stream>>>(brp, l2i, img_shape, refuv, maskf, NQ);
    valid_count_kernel<<<grid1(NQ), T, 0, stream>>>(maskf, validf, countf, NQ);

    // 4) GEMMs on the query side (computed ONCE, not per camera: q_rb is a broadcast)
    gemm_bf16_tile<<<gemm_grid(MQ, 256), 128, LDSB, stream>>>(
        qsum, WT_samp, b_samp, off_f, nullptr, nullptr, MQ, 256, CDIM);
    gemm_bf16_tile<<<gemm_grid(MQ, 128), 128, LDSB, stream>>>(
        qsum, WT_attn, b_attn, aw_f, nullptr, nullptr, MQ, 128, CDIM);
    softmax_kernel<<<grid1(NQ * HDS), T, 0, stream>>>(aw_f, NQ);

    // 5) value projection (bf16 out for cheap gathers)
    gemm_bf16_tile<<<gemm_grid(MV, CDIM), 128, LDSB, stream>>>(
        val_bf, WT_val, b_val, nullptr, vproj, nullptr, MV, CDIM, CDIM);

    // 6) deformable bilinear sampling -> attn_in bf16
    {
        int waves = CAMS * NQ * HDS;                   // one wave per (cam,q,head)
        int blocks = (waves * 32 + T - 1) / T;
        sample_kernel<<<blocks, T, 0, stream>>>(vproj, refuv, off_f, aw_f, shapes, lsi, attn_in, NQ, NK);
    }

    // 7) output projection per camera slot
    gemm_bf16_tile<<<gemm_grid(MA, CDIM), 128, LDSB, stream>>>(
        attn_in, WT_dout, b_dout, attn_out, nullptr, nullptr, MA, CDIM, CDIM);

    // 8) masked mean over cameras
    reduce_slots_kernel<<<grid1(NQ * CDIM), T, 0, stream>>>(attn_out, validf, countf, slots, NQ);

    // 9) final projection + residual, straight into d_out
    gemm_bf16_tile<<<gemm_grid(MQ, CDIM), 128, LDSB, stream>>>(
        slots, WT_out, b_out, (float*)d_out, nullptr, query, MQ, CDIM, CDIM);

    (void)n_in; (void)out_size; (void)ws_size;
}
